// DiT_Attention_51041391345891
// MI455X (gfx1250) — matrix operand order repaired and checked by reading, not compiled, not run on hardware
//
#include <hip/hip_runtime.h>
#include <hip/hip_bf16.h>

// ---------------- problem constants (fixed by the reference) ----------------
static constexpr int kB  = 2;
static constexpr int kS  = 2048;
static constexpr int kD  = 1024;
static constexpr int kH  = 16;
static constexpr int kHD = 64;

typedef __bf16 bf16;
typedef __attribute__((ext_vector_type(16))) __bf16 v16bf;
typedef __attribute__((ext_vector_type(8)))  __bf16 v8bf;
typedef __attribute__((ext_vector_type(8)))  float  v8f;

static __device__ __forceinline__ v8f wmma_bf16(v16bf a, v16bf b, v8f c) {
  // D = A(16x32 bf16) x B(32x16 bf16) + C(16x16 f32)
  return __builtin_amdgcn_wmma_f32_16x16x32_bf16(false, a, false, b, (short)0, c,
                                                 false, false);
}

static __device__ __forceinline__ v8f zero8() {
  v8f z = {0.f, 0.f, 0.f, 0.f, 0.f, 0.f, 0.f, 0.f};
  return z;
}

// ---- A-tile (16x32, M x K) loader from row-major bf16, row stride ld elems.
// ISA layout: lanes 0-15 row M=lane, elems 0..7 -> K=0..7, 8..15 -> K=16..23;
// lanes 16-31 same rows, K offset +8.  => two contiguous 16B loads per lane.
static __device__ __forceinline__ v16bf load_a_bf16(const bf16* base, int ld) {
  const int lane = threadIdx.x & 31;
  const int row  = lane & 15;
  const int koff = (lane & 16) ? 8 : 0;
  const v8bf lo = *(const v8bf*)(base + (size_t)row * ld + koff);
  const v8bf hi = *(const v8bf*)(base + (size_t)row * ld + koff + 16);
  v16bf a;
#pragma unroll
  for (int i = 0; i < 8; ++i) { a[i] = lo[i]; a[i + 8] = hi[i]; }
  return a;
}

// ---- A-tile loader from row-major f32 with scale (converts to bf16).
static __device__ __forceinline__ v16bf load_a_f32(const float* base, int ld,
                                                   float scale) {
  const int lane = threadIdx.x & 31;
  const int row  = lane & 15;
  const int koff = (lane & 16) ? 8 : 0;
  const float4* p0 = (const float4*)(base + (size_t)row * ld + koff);
  const float4* p1 = (const float4*)(base + (size_t)row * ld + koff + 16);
  float4 x0 = p0[0], x1 = p0[1], y0 = p1[0], y1 = p1[1];
  v16bf a;
  a[0]  = (bf16)(x0.x * scale); a[1]  = (bf16)(x0.y * scale);
  a[2]  = (bf16)(x0.z * scale); a[3]  = (bf16)(x0.w * scale);
  a[4]  = (bf16)(x1.x * scale); a[5]  = (bf16)(x1.y * scale);
  a[6]  = (bf16)(x1.z * scale); a[7]  = (bf16)(x1.w * scale);
  a[8]  = (bf16)(y0.x * scale); a[9]  = (bf16)(y0.y * scale);
  a[10] = (bf16)(y0.z * scale); a[11] = (bf16)(y0.w * scale);
  a[12] = (bf16)(y1.x * scale); a[13] = (bf16)(y1.y * scale);
  a[14] = (bf16)(y1.z * scale); a[15] = (bf16)(y1.w * scale);
  return a;
}

// ---- B-tile (32x16, K x N) loader. Memory holds B "N-major": element (n,k)
// at base + n*ld + k.  ISA: col N = lane&15; lanes 0-15 K=0..7,16..23, lanes 16-31
// K=8..15,24..31 (same order as A) => two 16B loads.
static __device__ __forceinline__ v16bf load_bt_bf16(const bf16* base, int ld) {
  const int lane = threadIdx.x & 31;
  const int col  = lane & 15;
  const int koff = (lane & 16) ? 8 : 0;
  const v8bf lo = *(const v8bf*)(base + (size_t)col * ld + koff);
  const v8bf hi = *(const v8bf*)(base + (size_t)col * ld + koff + 16);
  v16bf b;
#pragma unroll
  for (int i = 0; i < 8; ++i) { b[i] = lo[i]; b[i + 8] = hi[i]; }
  return b;
}

// ---- C/D tile (16x16 f32): VGPR g, lanes 0-15 row=g, lanes 16-31 row=g+8,
// col = lane&15.
static __device__ __forceinline__ void store_c_bf16(bf16* base, int ld, v8f c) {
  const int lane = threadIdx.x & 31;
  const int col  = lane & 15;
  const int r0   = (lane & 16) ? 8 : 0;
#pragma unroll
  for (int g = 0; g < 8; ++g)
    base[(size_t)(r0 + g) * ld + col] = (bf16)c[g];
}

static __device__ __forceinline__ void store_c_bf16_T(bf16* base, int ldT, v8f c) {
  const int lane = threadIdx.x & 31;
  const int col  = lane & 15;
  const int r0   = (lane & 16) ? 8 : 0;
#pragma unroll
  for (int g = 0; g < 8; ++g)
    base[(size_t)col * ldT + r0 + g] = (bf16)c[g];
}

// ======================= kernel 0: f32 -> bf16 convert ======================
__global__ void cvt_f32_bf16(const float* __restrict__ s, bf16* __restrict__ d,
                             int n) {
  int i = blockIdx.x * blockDim.x + threadIdx.x;
  if (i < n) d[i] = (bf16)s[i];
}

// ============ kernel 1: fused per-head projections (q1,k1,q2,k2,vT) =========
// grid (B*H, S/64), block 128 (4 waves, 16 rows each). 40 WMMA per wave.
// q-side scale folds both 1/sqrt(64) and log2(e) so softmax can use v_exp_f32
// in base-2 directly (softmax is invariant under the consistent rescale).
__global__ __launch_bounds__(128) void proj_kernel(
    const float* __restrict__ q, const float* __restrict__ k,
    const float* __restrict__ v, const bf16* __restrict__ wq1b,
    const bf16* __restrict__ wk1b, const bf16* __restrict__ wq2b,
    const bf16* __restrict__ wk2b, const bf16* __restrict__ wvb,
    bf16* __restrict__ q1o, bf16* __restrict__ k1o, bf16* __restrict__ q2o,
    bf16* __restrict__ k2o, bf16* __restrict__ vTo) {
  const int bh = blockIdx.x;            // b*H + h
  const int b  = bh / kH, h = bh % kH;
  const int wave = threadIdx.x >> 5;
  const int s0 = blockIdx.y * 64 + wave * 16;

  const float kQScale = 0.125f * 1.44269504088896340736f;  // (1/sqrt(64))*log2(e)
  const size_t xoff = (size_t)(b * kS + s0) * kD + h * kHD;
  v16bf aq[2], ak[2], av[2];
#pragma unroll
  for (int kh = 0; kh < 2; ++kh) {
    aq[kh] = load_a_f32(q + xoff + kh * 32, kD, kQScale);
    ak[kh] = load_a_f32(k + xoff + kh * 32, kD, 1.0f);
    av[kh] = load_a_f32(v + xoff + kh * 32, kD, 1.0f);
  }

#pragma unroll
  for (int nt = 0; nt < 4; ++nt) {
    v8f c0 = zero8(), c1 = zero8(), c2 = zero8(), c3 = zero8(), c4 = zero8();
#pragma unroll
    for (int kh = 0; kh < 2; ++kh) {
      const int woff = h * kHD * kHD + nt * 16 * kHD + kh * 32;
      c0 = wmma_bf16(aq[kh], load_bt_bf16(wq1b + woff, kHD), c0);
      c1 = wmma_bf16(ak[kh], load_bt_bf16(wk1b + woff, kHD), c1);
      c2 = wmma_bf16(aq[kh], load_bt_bf16(wq2b + woff, kHD), c2);
      c3 = wmma_bf16(ak[kh], load_bt_bf16(wk2b + woff, kHD), c3);
      c4 = wmma_bf16(av[kh], load_bt_bf16(wvb  + woff, kHD), c4);
    }
    const size_t obase = ((size_t)bh * kS + s0) * kHD + nt * 16;
    store_c_bf16(q1o + obase, kHD, c0);
    store_c_bf16(k1o + obase, kHD, c1);
    store_c_bf16(q2o + obase, kHD, c2);
    store_c_bf16(k2o + obase, kHD, c3);
    // vp stored transposed [b,h,e,s] so PV B-tiles are contiguous
    store_c_bf16_T(vTo + ((size_t)bh * kHD + nt * 16) * kS + s0, kS, c4);
  }
}

// =================== kernel 2: dual flash attention ========================
static __device__ __forceinline__ void softmax_pv(v8f s[2], float m[8],
                                                  float l[8], v8f o[4],
                                                  const v16bf bv[4],
                                                  bf16 (*ldsrow)[32]) {
  const int lane = threadIdx.x & 31;
  const int col  = lane & 15;
  const int r0   = (lane & 16) ? 8 : 0;
#pragma unroll
  for (int g = 0; g < 8; ++g) {
    float t = fmaxf(s[0][g], s[1][g]);
#pragma unroll
    for (int msk = 1; msk < 16; msk <<= 1)       // reduce within 16-lane half
      t = fmaxf(t, __shfl_xor(t, msk, 32));
    const float nm = fmaxf(m[g], t);
    const float p0 = __builtin_amdgcn_exp2f(s[0][g] - nm);  // base-2 domain
    const float p1 = __builtin_amdgcn_exp2f(s[1][g] - nm);
    float rs = p0 + p1;
#pragma unroll
    for (int msk = 1; msk < 16; msk <<= 1) rs += __shfl_xor(rs, msk, 32);
    const float corr = __builtin_amdgcn_exp2f(m[g] - nm);
    l[g] = l[g] * corr + rs;
    m[g] = nm;
#pragma unroll
    for (int nt = 0; nt < 4; ++nt) o[nt][g] *= corr;
    ldsrow[r0 + g][col]      = (bf16)p0;
    ldsrow[r0 + g][col + 16] = (bf16)p1;
  }
  asm volatile("s_wait_dscnt 0" ::: "memory");
  const v16bf pa = load_a_bf16(&ldsrow[0][0], 32);
#pragma unroll
  for (int nt = 0; nt < 4; ++nt) o[nt] = wmma_bf16(pa, bv[nt], o[nt]);
}

// grid (B*H, S/128), block 256 (8 waves, 16 q-rows each), 32 keys/iter.
// K1/K2/V^T tiles for the whole block are staged in LDS via async DMA
// (GLOBAL_LOAD_ASYNC_TO_LDS_B128, ASYNCcnt), double-buffered so the DMA of
// tile j+1 overlaps the WMMA work on tile j.
__global__ __launch_bounds__(256) void attn_kernel(
    const bf16* __restrict__ q1, const bf16* __restrict__ k1,
    const bf16* __restrict__ q2, const bf16* __restrict__ k2,
    const bf16* __restrict__ vT, const float* __restrict__ gainp,
    float* __restrict__ obuf) {
  __shared__ __align__(16) bf16 sk1[2][32][64];   // [buf][j][e]  8KB
  __shared__ __align__(16) bf16 sk2[2][32][64];   // [buf][j][e]  8KB
  __shared__ __align__(16) bf16 svT[2][64][32];   // [buf][e][j]  8KB
  __shared__ __align__(16) bf16 lds_p[8][16][32]; // wave-private P tiles 8KB

  const int bh = blockIdx.x;
  const int b  = bh / kH, h = bh % kH;
  const int wave = threadIdx.x >> 5;
  const int t = threadIdx.x;
  const int i0 = blockIdx.y * 128 + wave * 16;
  const size_t hb = (size_t)bh * kS * kHD;

  // Issue the block's shared K/V tile DMA for key rows [j0, j0+32).
  // k-tiles are 4KB contiguous (32 rows x 64 e); vT tile is 64 rows x 64B.
  auto issue_tiles = [&](int buf, int j0) {
    {
      unsigned long long g =
          (unsigned long long)(k1 + hb + (size_t)j0 * kHD) + (unsigned)t * 16u;
      unsigned l = (unsigned)(size_t)(&sk1[buf][0][0]) + (unsigned)t * 16u;
      asm volatile("global_load_async_to_lds_b128 %0, %1, off"
                   :: "v"(l), "v"(g) : "memory");
    }
    {
      unsigned long long g =
          (unsigned long long)(k2 + hb + (size_t)j0 * kHD) + (unsigned)t * 16u;
      unsigned l = (unsigned)(size_t)(&sk2[buf][0][0]) + (unsigned)t * 16u;
      asm volatile("global_load_async_to_lds_b128 %0, %1, off"
                   :: "v"(l), "v"(g) : "memory");
    }
    {
      const int e = t >> 2, part = t & 3;
      unsigned long long g =
          (unsigned long long)(vT + ((size_t)bh * kHD + e) * kS + j0) +
          (unsigned)part * 16u;
      unsigned l = (unsigned)(size_t)(&svT[buf][e][0]) + (unsigned)part * 16u;
      asm volatile("global_load_async_to_lds_b128 %0, %1, off"
                   :: "v"(l), "v"(g) : "memory");
    }
  };

  v16bf aq1[2], aq2[2];
#pragma unroll
  for (int kh = 0; kh < 2; ++kh) {
    aq1[kh] = load_a_bf16(q1 + hb + (size_t)i0 * kHD + kh * 32, kHD);
    aq2[kh] = load_a_bf16(q2 + hb + (size_t)i0 * kHD + kh * 32, kHD);
  }

  v8f o1[4], o2[4];
  float m1[8], l1[8], m2[8], l2[8];
#pragma unroll
  for (int nt = 0; nt < 4; ++nt) { o1[nt] = zero8(); o2[nt] = zero8(); }
#pragma unroll
  for (int g = 0; g < 8; ++g) {
    m1[g] = -__builtin_inff(); m2[g] = -__builtin_inff();
    l1[g] = 0.f;               l2[g] = 0.f;
  }

  // prologue: fill buffer 0
  issue_tiles(0, 0);
  asm volatile("s_wait_asynccnt 0" ::: "memory");
  __syncthreads();

  for (int it = 0; it < kS / 32; ++it) {
    const int cur = it & 1, nxt = cur ^ 1;
    const int j0 = it * 32;
    if (j0 + 32 < kS) issue_tiles(nxt, j0 + 32);  // overlap DMA with math

    v8f s1[2], s2[2];
#pragma unroll
    for (int ct = 0; ct < 2; ++ct) {
      s1[ct] = zero8(); s2[ct] = zero8();
#pragma unroll
      for (int kh = 0; kh < 2; ++kh) {
        s1[ct] = wmma_bf16(aq1[kh],
                           load_bt_bf16(&sk1[cur][ct * 16][kh * 32], kHD), s1[ct]);
        s2[ct] = wmma_bf16(aq2[kh],
                           load_bt_bf16(&sk2[cur][ct * 16][kh * 32], kHD), s2[ct]);
      }
    }
    v16bf bv[4];
#pragma unroll
    for (int nt = 0; nt < 4; ++nt)
      bv[nt] = load_bt_bf16(&svT[cur][nt * 16][0], 32);

    softmax_pv(s1, m1, l1, o1, bv, lds_p[wave]);
    softmax_pv(s2, m2, l2, o2, bv, lds_p[wave]);

    asm volatile("s_wait_asynccnt 0" ::: "memory");  // next buffer ready
    __syncthreads();
  }

  const float gain = *gainp;
  float w1[8], w2[8];
#pragma unroll
  for (int g = 0; g < 8; ++g) { w1[g] = 1.f / l1[g]; w2[g] = gain / l2[g]; }
  const int lane = threadIdx.x & 31;
  const int col  = lane & 15;
  const int r0   = (lane & 16) ? 8 : 0;
#pragma unroll
  for (int nt = 0; nt < 4; ++nt)
#pragma unroll
    for (int g = 0; g < 8; ++g) {
      const int srow = i0 + r0 + g;
      obuf[(((size_t)b * kS + srow) * kH + h) * kHD + nt * 16 + col] =
          o1[nt][g] * w1[g] - o2[nt][g] * w2[g];
    }
}

// ============== kernel 3: group-norm over hd=64 per (b,s,h) ================
__global__ __launch_bounds__(256) void gnorm_kernel(
    const float* __restrict__ obuf, bf16* __restrict__ xb) {
  const int wave = threadIdx.x >> 5, lane = threadIdx.x & 31;
  const size_t row = (size_t)blockIdx.x * 8 + wave;   // over B*S*H rows
  const float* p = obuf + row * kHD;
  const float x0 = p[lane], x1 = p[lane + 32];
  float sm = x0 + x1, sq = x0 * x0 + x1 * x1;
#pragma unroll
  for (int msk = 1; msk < 32; msk <<= 1) {
    sm += __shfl_xor(sm, msk, 32);
    sq += __shfl_xor(sq, msk, 32);
  }
  const float mean = sm * (1.f / 64.f);
  const float var  = sq * (1.f / 64.f) - mean * mean;
  const float rstd = rsqrtf(var + 1e-5f);
  bf16* o = xb + row * kHD;
  o[lane]      = (bf16)((x0 - mean) * rstd);
  o[lane + 32] = (bf16)((x1 - mean) * rstd);
}

// ================= kernel 4: output projection GEMM ========================
// out[4096,1024] = X(bf16)[4096,1024] @ wo^T ; grid (32,16), block 256.
__global__ __launch_bounds__(256) void outproj_kernel(
    const bf16* __restrict__ xb, const bf16* __restrict__ wob,
    float* __restrict__ out) {
  const int wave = threadIdx.x >> 5;
  const int m0 = blockIdx.x * 128 + wave * 16;
  const int n0 = blockIdx.y * 64;
  v8f acc[4];
#pragma unroll
  for (int nt = 0; nt < 4; ++nt) acc[nt] = zero8();
  for (int kk = 0; kk < kD; kk += 32) {
    const v16bf a = load_a_bf16(xb + (size_t)m0 * kD + kk, kD);
#pragma unroll
    for (int nt = 0; nt < 4; ++nt)
      acc[nt] = wmma_bf16(a, load_bt_bf16(wob + (size_t)(n0 + nt * 16) * kD + kk, kD),
                          acc[nt]);
  }
  const int lane = threadIdx.x & 31;
  const int col  = lane & 15;
  const int r0   = (lane & 16) ? 8 : 0;
#pragma unroll
  for (int nt = 0; nt < 4; ++nt)
#pragma unroll
    for (int g = 0; g < 8; ++g)
      out[(size_t)(m0 + r0 + g) * kD + n0 + nt * 16 + col] = acc[nt][g];
}

// ============================== launcher ===================================
extern "C" void kernel_launch(void* const* d_in, const int* in_sizes, int n_in,
                              void* d_out, int out_size, void* d_ws,
                              size_t ws_size, hipStream_t stream) {
  const float* q    = (const float*)d_in[0];
  const float* k    = (const float*)d_in[1];
  const float* v    = (const float*)d_in[2];
  const float* wq1  = (const float*)d_in[3];
  const float* wk1  = (const float*)d_in[4];
  const float* wq2  = (const float*)d_in[5];
  const float* wk2  = (const float*)d_in[6];
  const float* wv   = (const float*)d_in[7];
  const float* wo   = (const float*)d_in[8];
  const float* gain = (const float*)d_in[9];
  float* out = (float*)d_out;

  char* ws = (char*)d_ws;
  size_t off = 0;
  auto alloc = [&](size_t bytes) {
    void* p = ws + off;
    off += (bytes + 255) & ~(size_t)255;
    return p;
  };
  const size_t wsz  = (size_t)kH * kHD * kHD;          // per head-weight elems
  const size_t actz = (size_t)kB * kH * kS * kHD;      // per activation elems
  bf16* wq1b = (bf16*)alloc(wsz * 2);
  bf16* wk1b = (bf16*)alloc(wsz * 2);
  bf16* wq2b = (bf16*)alloc(wsz * 2);
  bf16* wk2b = (bf16*)alloc(wsz * 2);
  bf16* wvb  = (bf16*)alloc(wsz * 2);
  bf16* wob  = (bf16*)alloc((size_t)kD * kD * 2);
  bf16* q1b  = (bf16*)alloc(actz * 2);
  bf16* k1b  = (bf16*)alloc(actz * 2);
  bf16* q2b  = (bf16*)alloc(actz * 2);
  bf16* k2b  = (bf16*)alloc(actz * 2);
  bf16* vTb  = (bf16*)alloc(actz * 2);
  float* obuf = (float*)alloc(actz * 4);
  bf16* xb   = (bf16*)alloc(actz * 2);

  const int nw = (int)wsz;
  cvt_f32_bf16<<<(nw + 255) / 256, 256, 0, stream>>>(wq1, wq1b, nw);
  cvt_f32_bf16<<<(nw + 255) / 256, 256, 0, stream>>>(wk1, wk1b, nw);
  cvt_f32_bf16<<<(nw + 255) / 256, 256, 0, stream>>>(wq2, wq2b, nw);
  cvt_f32_bf16<<<(nw + 255) / 256, 256, 0, stream>>>(wk2, wk2b, nw);
  cvt_f32_bf16<<<(nw + 255) / 256, 256, 0, stream>>>(wv, wvb, nw);
  cvt_f32_bf16<<<(kD * kD + 255) / 256, 256, 0, stream>>>(wo, wob, kD * kD);

  proj_kernel<<<dim3(kB * kH, kS / 64), 128, 0, stream>>>(
      q, k, v, wq1b, wk1b, wq2b, wk2b, wvb, q1b, k1b, q2b, k2b, vTb);

  attn_kernel<<<dim3(kB * kH, kS / 128), 256, 0, stream>>>(
      q1b, k1b, q2b, k2b, vTb, gain, obuf);

  gnorm_kernel<<<(kB * kS * kH) / 8, 256, 0, stream>>>(obuf, xb);

  outproj_kernel<<<dim3((kB * kS) / 128, kD / 64), 256, 0, stream>>>(
      xb, wob, out);
}
